// NeighborAttent_65910568125056
// MI455X (gfx1250) — matrix operand order repaired and checked by reading, not compiled, not run on hardware
//
#include <hip/hip_runtime.h>
#include <hip/hip_bf16.h>

#define BB  4
#define TT  1024
#define CC  128
#define NEI 4
#define NEGV -1e22f

typedef __attribute__((ext_vector_type(16))) __bf16 v16bf;
typedef __attribute__((ext_vector_type(8)))  __bf16 v8bf;
typedef __attribute__((ext_vector_type(8)))  float  v8f;

// LDS layout of attn_kernel (dynamic shared memory, base offset 0)
#define LDS_SCORE 0        // float  [16][1024]  65536 B
#define LDS_QSI   65536    // __bf16 [16*128]     4096 B
#define LDS_QNI   69632    // __bf16 [64*128]    16384 B
#define LDS_KNI   86016    // __bf16 [64*128]    16384 B
#define LDS_P     102400   // __bf16 [16][1024]  32768 B
#define SMEM_BYTES 135168

// ---------------------------------------------------------------------------
// WMMA helpers (gfx1250 wave32, 16x16x32 bf16, f32 accum)
// ---------------------------------------------------------------------------
__device__ __forceinline__ v8f wmma_bf16(v16bf a, v16bf b, v8f c) {
  return __builtin_amdgcn_wmma_f32_16x16x32_bf16(false, a, false, b, (short)0, c,
                                                 false, false);
}

// A-matrix 16x32 bf16 fragment from a row-major matrix (row stride ld halves).
// Lane l (lo=l&15,h=l>>4) holds row lo; h=0 -> K {0..7,16..23}, h=1 -> {8..15,24..31}.
__device__ __forceinline__ v16bf load_frag_A_bf(const __bf16* base, int ld, int kc,
                                                int lane) {
  const int lo = lane & 15, h = lane >> 4;
  const __bf16* p = base + (size_t)lo * ld + kc * 32;
  v16bf f;
  *(v8bf*)&f       = *(const v8bf*)(p + h * 8);
  *((v8bf*)&f + 1) = *(const v8bf*)(p + 16 + h * 8);
  return f;
}

// B-matrix 32x16 bf16 fragment: column n = row (base+n) of the source matrix
// (i.e. computes X * Yt when both sides are row-major gram inputs).
// Lane l (h=l>>4): h=0 -> K {0..7,16..23}, h=1 -> {8..15,24..31}.
__device__ __forceinline__ v16bf load_frag_B_bf(const __bf16* base, int ld, int kc,
                                                int lane) {
  const int lo = lane & 15, h = lane >> 4;
  const __bf16* p = base + (size_t)lo * ld + kc * 32 + h * 8;
  v16bf f;
  *(v8bf*)&f       = *(const v8bf*)(p);
  *((v8bf*)&f + 1) = *(const v8bf*)(p + 16);
  return f;
}

// async global -> LDS 16-byte copy (per lane), tracked by ASYNCcnt
__device__ __forceinline__ void async_copy16(unsigned lds_off, const void* gaddr) {
  asm volatile("global_load_async_to_lds_b128 %0, %1, off"
               :: "v"(lds_off), "v"(gaddr) : "memory");
}
__device__ __forceinline__ void wait_async0() {
  asm volatile("s_wait_asynccnt 0x0" ::: "memory");
}

// ---------------------------------------------------------------------------
// K0: recover neighbor indices from one-hot inxs (B,T,NEI,T). One wave / row.
// ---------------------------------------------------------------------------
__global__ __launch_bounds__(128) void idx_kernel(const float* __restrict__ inxs,
                                                  int* __restrict__ idx) {
  const int w = threadIdx.x >> 5, lane = threadIdx.x & 31;
  const int row = blockIdx.x * 4 + w;  // B*T*NEI rows
  const float* p = inxs + (size_t)row * TT;
  int found = 0;
  for (int c = lane; c < TT; c += 32)
    if (p[c] > 0.5f) found = c;
#pragma unroll
  for (int o = 16; o; o >>= 1) {
    int v = __shfl_xor(found, o, 32);
    found = found > v ? found : v;
  }
  if (lane == 0) idx[row] = found;
}

// ---------------------------------------------------------------------------
// K1: projections + L2 norm.  q,k,qs -> bf16 rows;  v -> bf16 transposed (C,T).
// One block per (b,t) row, 128 threads (one per output channel).
// ---------------------------------------------------------------------------
__global__ __launch_bounds__(128) void proj_kernel(
    const float* __restrict__ x, const float* __restrict__ Wq,
    const float* __restrict__ Wk, const float* __restrict__ Wqs,
    const float* __restrict__ Wv, __bf16* __restrict__ qb,
    __bf16* __restrict__ kb, __bf16* __restrict__ qsb,
    __bf16* __restrict__ vtb) {
  __shared__ float xr[CC];
  __shared__ float red[3][CC];
  const int bt = blockIdx.x;      // b*T + t
  const int b  = bt >> 10, t = bt & 1023;
  const int co = threadIdx.x;
  xr[co] = x[(size_t)bt * CC + co];
  __syncthreads();

  const float* wq  = Wq  + (size_t)co * CC;
  const float* wk  = Wk  + (size_t)co * CC;
  const float* wqs = Wqs + (size_t)co * CC;
  const float* wv  = Wv  + (size_t)co * CC;
  float aq = 0.f, ak = 0.f, aqs = 0.f, av = 0.f;
#pragma unroll 8
  for (int ci = 0; ci < CC; ci++) {
    const float xv = xr[ci];
    aq  += xv * wq[ci];
    ak  += xv * wk[ci];
    aqs += xv * wqs[ci];
    av  += xv * wv[ci];
  }
  red[0][co] = aq * aq;
  red[1][co] = ak * ak;
  red[2][co] = aqs * aqs;
  __syncthreads();
  for (int s = 64; s > 0; s >>= 1) {
    if (co < s) {
      red[0][co] += red[0][co + s];
      red[1][co] += red[1][co + s];
      red[2][co] += red[2][co + s];
    }
    __syncthreads();
  }
  const float iq  = 1.f / fmaxf(sqrtf(red[0][0]), 1e-12f);
  const float ik  = 1.f / fmaxf(sqrtf(red[1][0]), 1e-12f);
  const float iqs = 1.f / fmaxf(sqrtf(red[2][0]), 1e-12f);
  qb [(size_t)bt * CC + co] = (__bf16)(aq * iq);
  kb [(size_t)bt * CC + co] = (__bf16)(ak * ik);
  qsb[(size_t)bt * CC + co] = (__bf16)(aqs * iqs);
  vtb[((size_t)b * CC + co) * TT + t] = (__bf16)av;  // transposed V
}

// ---------------------------------------------------------------------------
// K2: gather qn/kn rows (B*T*NEI, C) bf16 so WMMA B-fragments are contiguous.
// ---------------------------------------------------------------------------
__global__ __launch_bounds__(64) void gather_kernel(
    const __bf16* __restrict__ qb, const __bf16* __restrict__ kb,
    const int* __restrict__ idx, __bf16* __restrict__ qnb,
    __bf16* __restrict__ knb) {
  const int row = blockIdx.x;           // (b*T + t)*NEI + n
  const int tth = threadIdx.x;          // 64 threads -> 64 uints = 128 bf16
  const int b = row / (TT * NEI);
  const int src = idx[row];
  const unsigned* qs = (const unsigned*)(qb + ((size_t)b * TT + src) * CC);
  const unsigned* ks = (const unsigned*)(kb + ((size_t)b * TT + src) * CC);
  unsigned* qd = (unsigned*)(qnb + (size_t)row * CC);
  unsigned* kd = (unsigned*)(knb + (size_t)row * CC);
  qd[tth] = qs[tth];
  kd[tth] = ks[tth];
}

// ---------------------------------------------------------------------------
// K3: fused neighbor attention.  1 block = (b, 16 query rows), 4 waves.
// ---------------------------------------------------------------------------
__global__ __launch_bounds__(128) void attn_kernel(
    const float* __restrict__ adj, const __bf16* __restrict__ qb,
    const __bf16* __restrict__ kb, const __bf16* __restrict__ qsb,
    const __bf16* __restrict__ qnb, const __bf16* __restrict__ knb,
    const __bf16* __restrict__ vtb, const float* __restrict__ w_nn,
    const float* __restrict__ w_ns, float* __restrict__ out) {
  extern __shared__ __align__(16) char smem[];
  float (*score)[TT] = (float(*)[TT])(smem + LDS_SCORE);
  __bf16* qsi = (__bf16*)(smem + LDS_QSI);
  __bf16* qni = (__bf16*)(smem + LDS_QNI);
  __bf16* kni = (__bf16*)(smem + LDS_KNI);
  __bf16 (*P)[TT] = (__bf16(*)[TT])(smem + LDS_P);

  const int b  = blockIdx.x >> 6;
  const int i0 = (blockIdx.x & 63) * 16;
  const int tid = threadIdx.x, w = tid >> 5, lane = tid & 31;
  const int lo = lane & 15, h = lane >> 4;

  const float wn0 = w_nn[0], wn1 = w_nn[1], wn2 = w_nn[2], wn3 = w_nn[3];
  const float ws0 = w_ns[0], ws1 = w_ns[1], ws2 = w_ns[2], ws3 = w_ns[3];

  // ---- stage A-side tiles into LDS via async global->LDS (ASYNCcnt) ----
  {
    const char* gq = (const char*)(qsb + ((size_t)b * TT + i0) * CC);        // 4 KB
    const char* gn = (const char*)(qnb + ((size_t)b * TT + i0) * NEI * CC);  // 16 KB
    const char* gk = (const char*)(knb + ((size_t)b * TT + i0) * NEI * CC);  // 16 KB
    for (int i = tid; i < 256; i += 128)
      async_copy16(LDS_QSI + i * 16, gq + (size_t)i * 16);
    for (int i = tid; i < 1024; i += 128) {
      async_copy16(LDS_QNI + i * 16, gn + (size_t)i * 16);
      async_copy16(LDS_KNI + i * 16, gk + (size_t)i * 16);
    }
    wait_async0();
  }
  __syncthreads();

  // ---- score assembly: each wave owns j-tiles jt = w, w+4, ... ----
  for (int jt = w; jt < 64; jt += 4) {
    const int j0 = jt * 16;

    // sim_s = qs_i . qs_j   (+ adjacency mask)  -> initializes score cols
    {
      v8f acc = {};
#pragma unroll
      for (int kc = 0; kc < 4; kc++)
        acc = wmma_bf16(load_frag_A_bf(qsi, CC, kc, lane),
                        load_frag_B_bf(qsb + ((size_t)b * TT + j0) * CC, CC, kc, lane),
                        acc);
#pragma unroll
      for (int v = 0; v < 8; v++) {
        const int row = v + 8 * h;
        const float m =
            (adj[((size_t)b * TT + (i0 + row)) * TT + j0 + lo] == 0.f) ? NEGV : 0.f;
        score[row][j0 + lo] = acc[v] + m;
      }
    }

    // sim_ns = sum_n w_ns[n] * (kn_i[n] . k_j)
    {
      v16bf bk[4];
#pragma unroll
      for (int kc = 0; kc < 4; kc++)
        bk[kc] = load_frag_B_bf(kb + ((size_t)b * TT + j0) * CC, CC, kc, lane);
#pragma unroll
      for (int ti2 = 0; ti2 < 4; ti2++) {
        v8f acc = {};
#pragma unroll
        for (int kc = 0; kc < 4; kc++)
          acc = wmma_bf16(load_frag_A_bf(kni + ti2 * 16 * CC, CC, kc, lane),
                          bk[kc], acc);
        // row r = v+8h  ->  i_local = 4*ti2 + v/4 + 2h,  n = v%4
        const float s0 = ws0 * acc[0] + ws1 * acc[1] + ws2 * acc[2] + ws3 * acc[3];
        const float s1 = ws0 * acc[4] + ws1 * acc[5] + ws2 * acc[6] + ws3 * acc[7];
        score[4 * ti2 + 2 * h + 0][j0 + lo] += s0;
        score[4 * ti2 + 2 * h + 1][j0 + lo] += s1;
      }
    }

    // sim_nn = sum_n w_nn[n] * max_m (qn_i[n] . qn_j[m])
#pragma unroll
    for (int tj2 = 0; tj2 < 4; tj2++) {
      const __bf16* bbase = qnb + (((size_t)b * TT + j0) * NEI + 16 * tj2) * CC;
      v16bf bq[4];
#pragma unroll
      for (int kc = 0; kc < 4; kc++) bq[kc] = load_frag_B_bf(bbase, CC, kc, lane);
#pragma unroll
      for (int ti2 = 0; ti2 < 4; ti2++) {
        v8f acc = {};
#pragma unroll
        for (int kc = 0; kc < 4; kc++)
          acc = wmma_bf16(load_frag_A_bf(qni + ti2 * 16 * CC, CC, kc, lane),
                          bq[kc], acc);
        // max over m: columns are lanes; m = lane%4 -> xor-reduce over 1,2
        float pl[8];
#pragma unroll
        for (int v = 0; v < 8; v++) {
          float xv = acc[v];
          xv = fmaxf(xv, __shfl_xor(xv, 1, 32));
          xv = fmaxf(xv, __shfl_xor(xv, 2, 32));
          pl[v] = xv;
        }
        const float s0 = wn0 * pl[0] + wn1 * pl[1] + wn2 * pl[2] + wn3 * pl[3];
        const float s1 = wn0 * pl[4] + wn1 * pl[5] + wn2 * pl[6] + wn3 * pl[7];
        if ((lane & 3) == 0) {
          const int col = j0 + 4 * tj2 + (lo >> 2);
          score[4 * ti2 + 2 * h + 0][col] += s0;
          score[4 * ti2 + 2 * h + 1][col] += s1;
        }
      }
    }
  }
  __syncthreads();

  // ---- softmax over each of 16 rows (4 rows / wave); emit bf16 P tile ----
  for (int r = 4 * w; r < 4 * w + 4; r++) {
    float mx = -3.0e38f;
    for (int c = lane; c < TT; c += 32) mx = fmaxf(mx, score[r][c]);
#pragma unroll
    for (int o = 16; o; o >>= 1) mx = fmaxf(mx, __shfl_xor(mx, o, 32));
    float sum = 0.f;
    for (int c = lane; c < TT; c += 32) {
      const float e = __expf(score[r][c] - mx);
      score[r][c] = e;
      sum += e;
    }
#pragma unroll
    for (int o = 16; o; o >>= 1) sum += __shfl_xor(sum, o, 32);
    const float inv = 1.0f / sum;
    for (int c = lane; c < TT; c += 32) P[r][c] = (__bf16)(score[r][c] * inv);
  }
  __syncthreads();

  // ---- out = P @ V : M=16, K=1024, N=128 (each wave: 2 N-tiles of 16) ----
  for (int nt = 2 * w; nt < 2 * w + 2; nt++) {
    v8f acc = {};
    for (int kc = 0; kc < 32; kc++)
      acc = wmma_bf16(load_frag_A_bf(&P[0][0], TT, kc, lane),
                      load_frag_B_bf(vtb + ((size_t)b * CC + 16 * nt) * TT, TT, kc, lane),
                      acc);
    const int col = 16 * nt + lo;
#pragma unroll
    for (int v = 0; v < 8; v++)
      out[((size_t)b * TT + (i0 + v + 8 * h)) * CC + col] = acc[v];
  }
}

// ---------------------------------------------------------------------------
// host launcher
// ---------------------------------------------------------------------------
extern "C" void kernel_launch(void* const* d_in, const int* in_sizes, int n_in,
                              void* d_out, int out_size, void* d_ws, size_t ws_size,
                              hipStream_t stream) {
  (void)in_sizes; (void)n_in; (void)out_size; (void)ws_size;
  const float* x    = (const float*)d_in[0];
  const float* adj  = (const float*)d_in[1];
  const float* inxs = (const float*)d_in[2];
  const float* Wq   = (const float*)d_in[3];
  const float* Wk   = (const float*)d_in[4];
  const float* Wqs  = (const float*)d_in[5];
  const float* Wv   = (const float*)d_in[6];
  const float* w_nn = (const float*)d_in[7];
  const float* w_ns = (const float*)d_in[8];
  float* out = (float*)d_out;

  char* ws = (char*)d_ws;
  size_t off = 0;
  auto carve = [&](size_t bytes) -> void* {
    void* p = (void*)(ws + off);
    off += (bytes + 255) & ~(size_t)255;
    return p;
  };
  int*    idx = (int*)   carve((size_t)BB * TT * NEI * sizeof(int));
  __bf16* qb  = (__bf16*)carve((size_t)BB * TT * CC * 2);
  __bf16* kb  = (__bf16*)carve((size_t)BB * TT * CC * 2);
  __bf16* qsb = (__bf16*)carve((size_t)BB * TT * CC * 2);
  __bf16* vtb = (__bf16*)carve((size_t)BB * CC * TT * 2);
  __bf16* qnb = (__bf16*)carve((size_t)BB * TT * NEI * CC * 2);
  __bf16* knb = (__bf16*)carve((size_t)BB * TT * NEI * CC * 2);

  idx_kernel   <<<BB * TT * NEI / 4, 128, 0, stream>>>(inxs, idx);
  proj_kernel  <<<BB * TT,           128, 0, stream>>>(x, Wq, Wk, Wqs, Wv,
                                                       qb, kb, qsb, vtb);
  gather_kernel<<<BB * TT * NEI,      64, 0, stream>>>(qb, kb, idx, qnb, knb);
  attn_kernel  <<<BB * (TT / 16),    128, SMEM_BYTES, stream>>>(
      adj, qb, kb, qsb, qnb, knb, vtb, w_nn, w_ns, out);
}
